// ChannelWiseLinear_75436805587567
// MI455X (gfx1250) — matrix operand order repaired and checked by reading, not compiled, not run on hardware
//
#include <hip/hip_runtime.h>

// ---------------------------------------------------------------------------
// ChannelWiseLinear for MI455X (gfx1250, wave32, WMMA).
//   y[b,c,g] = sum_f x[b,c,f] * W[c,g,f] + bias[c,g]
// B=64, C=128, F=G=1024.
//
// 128 independent 64x1024x1024 GEMMs. fp32 is split in-kernel into bf16
// hi/lo (v = vh + vl) and computed with 3x v_wmma_f32_16x16x32_bf16 per
// tile-K (xh*wh + xh*wl + xl*wh): ~3x bf16 FLOPs stays under the
// 576MB / 23.3TB/s HBM floor (~25us) with near-fp32 accuracy.
//
// Both the W tile (128x32) and x tile (64x32) are staged per workgroup
// through LDS: every element is loaded + bf16-split exactly ONCE, stored
// hi/lo in fragment-ready layouts, and re-read by the 8 waves as
// conflict-free ds_load_b128 fragments. Double-buffered (48KB), one
// barrier per K-chunk. Staging global loads issue at the TOP of each
// iteration; the split+ds_store runs AFTER the WMMAs, so the WMMA window
// keeps all fragments resident and overlaps the 4KB-strided W fetches.
// ---------------------------------------------------------------------------

typedef __attribute__((ext_vector_type(16))) __bf16 v16bf;
typedef __attribute__((ext_vector_type(8)))  float  v8f;

constexpr int Cdim = 128;
constexpr int Fdim = 1024;
constexpr int NBLK = 128;   // output columns per workgroup

union FragU {                 // 32B: one 16x-bf16 WMMA operand
    uint4  q[2];
    v16bf  v;
    __bf16 e[16];
};
union Pack8 {                 // 16B: 8 bf16
    uint4  q;
    __bf16 e[8];
};

__device__ __forceinline__ void split_bf16(float f, __bf16& h, __bf16& l) {
    __bf16 hh = (__bf16)f;              // RNE to bf16
    h = hh;
    l = (__bf16)(f - (float)hh);        // residual
}

__device__ __forceinline__ void split16(const float* wf, FragU& ph, FragU& pl) {
    #pragma unroll
    for (int i = 0; i < 16; ++i) split_bf16(wf[i], ph.e[i], pl.e[i]);
}

__device__ __forceinline__ void split8(const float* wf, Pack8& ph, Pack8& pl) {
    #pragma unroll
    for (int i = 0; i < 8; ++i) split_bf16(wf[i], ph.e[i], pl.e[i]);
}

__global__ __launch_bounds__(256)
void cwlinear_wmma_bf16x3(const float* __restrict__ X,   // [B, C, F]
                          const float* __restrict__ W,   // [C, F(g), F(f)]
                          const float* __restrict__ bias,// [C, F]
                          float* __restrict__ Y) {       // [B, C, F]
    // B tile: [buf][hi/lo][nt 0..7][16B chunk 0..1][lane 0..31]  -> 32 KB
    __shared__ uint4 ldsB[2][2][8][2][32];
    // A tile: [buf][hi/lo][mt 0..3][kchunk 0..3][row 0..15]      -> 16 KB
    __shared__ uint4 ldsA[2][2][4][4][16];

    const int bx   = blockIdx.x;
    const int c    = bx >> 3;                 // 8 column blocks per channel
    const int gblk = (bx & 7) * NBLK;

    const int tid    = threadIdx.x;
    const int lane   = tid & 31;
    const int wave   = tid >> 5;              // 0..7
    const int mt     = wave & 3;              // M tile (16 rows), covers B=64
    const int m0     = mt * 16;
    const int ntbase = (wave >> 2) * 4;       // 4 N tiles of 16 per wave

    const int ln = lane & 15;
    const int lh = lane >> 4;

    const float* wch = W + (size_t)c * Fdim * Fdim;

    // --- B staging: thread t owns W row sr, 16-float K-half sh -------------
    const int sr    = tid >> 1;               // 0..127
    const int sh    = tid & 1;
    const int snt   = sr >> 4;
    const int slane = (sr & 15) + 16 * sh;
    const float* srowB = wch + (size_t)(gblk + sr) * Fdim + 16 * sh;

    // --- A staging: thread t owns x row am, 8-float K-chunk akc ------------
    const int am  = tid & 63;                 // batch row 0..63
    const int akc = tid >> 6;                 // K chunk 0..3
    const int amt = am >> 4;
    const int aln = am & 15;
    const float* srowA = X + ((size_t)am * Cdim + c) * Fdim + 8 * akc;

    v8f acc[4] = {v8f{0.f,0.f,0.f,0.f,0.f,0.f,0.f,0.f},
                  v8f{0.f,0.f,0.f,0.f,0.f,0.f,0.f,0.f},
                  v8f{0.f,0.f,0.f,0.f,0.f,0.f,0.f,0.f},
                  v8f{0.f,0.f,0.f,0.f,0.f,0.f,0.f,0.f}};

    // Fragment loads + 12 WMMAs for the K-chunk staged in buffer bb.
    auto compute_chunk = [&](int bb) {
        FragU ah, al;
        ah.q[0] = ldsA[bb][0][mt][2 * lh][ln];
        ah.q[1] = ldsA[bb][0][mt][2 * lh + 1][ln];
        al.q[0] = ldsA[bb][1][mt][2 * lh][ln];
        al.q[1] = ldsA[bb][1][mt][2 * lh + 1][ln];
        FragU bh[4], bl[4];
        #pragma unroll
        for (int nl = 0; nl < 4; ++nl) {
            const int nt = ntbase + nl;
            bh[nl].q[0] = ldsB[bb][0][nt][0][lane];
            bh[nl].q[1] = ldsB[bb][0][nt][1][lane];
            bl[nl].q[0] = ldsB[bb][1][nt][0][lane];
            bl[nl].q[1] = ldsB[bb][1][nt][1][lane];
        }
        // y += xl*wh + xh*wl + xh*wh   (small terms first)
        #pragma unroll
        for (int nl = 0; nl < 4; ++nl) {
            acc[nl] = __builtin_amdgcn_wmma_f32_16x16x32_bf16(
                false, al.v, false, bh[nl].v, (short)0, acc[nl], false, false);
            acc[nl] = __builtin_amdgcn_wmma_f32_16x16x32_bf16(
                false, ah.v, false, bl[nl].v, (short)0, acc[nl], false, false);
            acc[nl] = __builtin_amdgcn_wmma_f32_16x16x32_bf16(
                false, ah.v, false, bh[nl].v, (short)0, acc[nl], false, false);
        }
    };

    // ---------------- prologue: stage K-chunk 0 into buf 0 ----------------
    {
        const float4* psb = (const float4*)(srowB);
        const float4 w0 = psb[0], w1 = psb[1], w2 = psb[2], w3 = psb[3];
        const float wf[16] = {w0.x,w0.y,w0.z,w0.w, w1.x,w1.y,w1.z,w1.w,
                              w2.x,w2.y,w2.z,w2.w, w3.x,w3.y,w3.z,w3.w};
        FragU ph, pl;
        split16(wf, ph, pl);
        ldsB[0][0][snt][0][slane] = ph.q[0];
        ldsB[0][0][snt][1][slane] = ph.q[1];
        ldsB[0][1][snt][0][slane] = pl.q[0];
        ldsB[0][1][snt][1][slane] = pl.q[1];

        const float4* psa = (const float4*)(srowA);
        const float4 a0 = psa[0], a1 = psa[1];
        const float af[8] = {a0.x,a0.y,a0.z,a0.w, a1.x,a1.y,a1.z,a1.w};
        Pack8 qh, ql;
        split8(af, qh, ql);
        ldsA[0][0][amt][akc][aln] = qh.q;
        ldsA[0][1][amt][akc][aln] = ql.q;
    }
    __syncthreads();

    // ------------------- steady state: chunks 0..30 ------------------------
    for (int kc = 0; kc < 31; ++kc) {
        const int bb = kc & 1;
        const int k1 = (kc + 1) * 32;

        // 1) issue next chunk's global loads EARLY (overlap with WMMAs).
        const float4* psb = (const float4*)(srowB + k1);
        const float4 w0 = psb[0], w1 = psb[1], w2 = psb[2], w3 = psb[3];
        const float4* psa = (const float4*)(srowA + k1);
        const float4 a0 = psa[0], a1 = psa[1];

        // 2+3) fragments from LDS buf bb, 12 WMMAs.
        compute_chunk(bb);

        // 4) split + store next chunk into the other buffer.
        {
            const float wf[16] = {w0.x,w0.y,w0.z,w0.w, w1.x,w1.y,w1.z,w1.w,
                                  w2.x,w2.y,w2.z,w2.w, w3.x,w3.y,w3.z,w3.w};
            FragU ph, pl;
            split16(wf, ph, pl);
            const int nb = bb ^ 1;
            ldsB[nb][0][snt][0][slane] = ph.q[0];
            ldsB[nb][0][snt][1][slane] = ph.q[1];
            ldsB[nb][1][snt][0][slane] = pl.q[0];
            ldsB[nb][1][snt][1][slane] = pl.q[1];

            const float af[8] = {a0.x,a0.y,a0.z,a0.w, a1.x,a1.y,a1.z,a1.w};
            Pack8 qh, ql;
            split8(af, qh, ql);
            ldsA[nb][0][amt][akc][aln] = qh.q;
            ldsA[nb][1][amt][akc][aln] = ql.q;
        }
        // 5) prefetch W two chunks ahead (global_prefetch_b8).
        if (kc < 30) __builtin_prefetch(srowB + k1 + 32, 0, 3);

        __syncthreads();
    }

    // --------------------------- last chunk (31) ---------------------------
    compute_chunk(1);

    // ------------------------------ epilogue -------------------------------
    // C/D layout: VGPR r of lane -> element (M = r + 8*lh, N = ln).
    #pragma unroll
    for (int nl = 0; nl < 4; ++nl) {
        const int   g  = gblk + (ntbase + nl) * 16 + ln;
        const float bv = bias[(size_t)c * Fdim + g];
        #pragma unroll
        for (int r = 0; r < 8; ++r) {
            const int m = m0 + r + 8 * lh;
            Y[((size_t)m * Cdim + c) * Fdim + g] = acc[nl][r] + bv;
        }
    }
}

extern "C" void kernel_launch(void* const* d_in, const int* in_sizes, int n_in,
                              void* d_out, int out_size, void* d_ws, size_t ws_size,
                              hipStream_t stream) {
    const float* X    = (const float*)d_in[0];   // [64,128,1024]
    const float* W    = (const float*)d_in[1];   // [128,1024,1024]
    const float* bias = (const float*)d_in[2];   // [128,1024]
    float*       Y    = (float*)d_out;           // [64,128,1024]

    const int nblocks = Cdim * (Fdim / NBLK);    // 128 * 8 = 1024 workgroups
    cwlinear_wmma_bf16x3<<<dim3(nblocks), dim3(256), 0, stream>>>(X, W, bias, Y);
}
